// GroupedQueryAttention_22651657519230
// MI455X (gfx1250) — matrix operand order repaired and checked by reading, not compiled, not run on hardware
//
#include <hip/hip_runtime.h>

// ---------------------------------------------------------------------------
// Types for CDNA5 WMMA (wave32, 16x16x32 bf16 -> f32)
// ---------------------------------------------------------------------------
typedef __bf16 v16bf __attribute__((ext_vector_type(16)));
typedef float  v8f   __attribute__((ext_vector_type(8)));
typedef unsigned int u32x4 __attribute__((ext_vector_type(4)));

union FragU { v16bf v; u32x4 q[2]; };

__device__ __forceinline__ __bf16 f2bf(float f) {
  unsigned u = __builtin_bit_cast(unsigned, f);
  u += 0x7FFFu + ((u >> 16) & 1u);                // round-to-nearest-even
  unsigned short h = (unsigned short)(u >> 16);
  return __builtin_bit_cast(__bf16, h);
}

// Load a 32-byte fragment as two 16-byte chunks (global or LDS pointer).
__device__ __forceinline__ v16bf load_frag2(const __bf16* p0, const __bf16* p1) {
  FragU f;
  f.q[0] = *(const u32x4*)p0;
  f.q[1] = *(const u32x4*)p1;
  return f.v;
}

__device__ __forceinline__ v8f wmma_bf16(v16bf a, v16bf b, v8f c) {
  return __builtin_amdgcn_wmma_f32_16x16x32_bf16(false, a, false, b, (short)0, c,
                                                 false, false);
}

// XOR-swizzle across lanes (group-of-32 mode, and_mask=0x1F).
template<int XM>
__device__ __forceinline__ float swz_xor(float x) {
  int i = __builtin_amdgcn_ds_swizzle(__builtin_bit_cast(int, x),
                                      (XM << 10) | 0x1F);
  return __builtin_bit_cast(float, i);
}

template<typename T> __device__ __forceinline__ T to_out(float v);
template<> __device__ __forceinline__ float  to_out<float >(float v) { return v; }
template<> __device__ __forceinline__ __bf16 to_out<__bf16>(float v) { return f2bf(v); }

// ---------------------------------------------------------------------------
// Conversion / layout kernels
// ---------------------------------------------------------------------------
__global__ __launch_bounds__(256) void k_cvt(const float* __restrict__ src,
                                             __bf16* __restrict__ dst, int n) {
  int i = blockIdx.x * blockDim.x + threadIdx.x;
  if (i < n) dst[i] = f2bf(src[i]);
}

// W [K][N] f32 row-major  ->  Wt [N][K] bf16 row-major
__global__ __launch_bounds__(256) void k_cvt_t(const float* __restrict__ src,
                                               __bf16* __restrict__ dst,
                                               int K, int N) {
  int i = blockIdx.x * blockDim.x + threadIdx.x;
  if (i < K * N) {
    int k = i / N, n = i - k * N;
    dst[(size_t)n * K + k] = f2bf(src[i]);
  }
}

// v [B*L][512] bf16 -> vt [B][8][64][L] bf16  (per-head transpose for P@V)
__global__ __launch_bounds__(256) void k_vT(const __bf16* __restrict__ v,
                                            __bf16* __restrict__ vt) {
  int i = blockIdx.x * blockDim.x + threadIdx.x;   // over 4096*512
  if (i < 4096 * 512) {
    int lg = i >> 9, c = i & 511;
    int b = lg >> 11, l = lg & 2047;
    int kvh = c >> 6, d = c & 63;
    vt[(((size_t)(b * 8 + kvh) * 64 + d) * 2048) + l] = v[i];
  }
}

// ---------------------------------------------------------------------------
// bf16 WMMA GEMM:  C[M][N] = A[M][K] * Bt[N][K]^T + bias[N]
// One wave computes a 64(M) x 64(N) tile: 4 A-frags x 4 B-frags -> 16 WMMAs
// per 32-wide K-step (1 b128 load per WMMA). M, N must be multiples of 64.
// ---------------------------------------------------------------------------
template<typename OutT>
__global__ __launch_bounds__(256)
void gemm_bf16_wmma(const __bf16* __restrict__ A, const __bf16* __restrict__ Bt,
                    const float* __restrict__ bias, OutT* __restrict__ C,
                    int M, int N, int K) {
  const int wave = (blockIdx.x * blockDim.x + threadIdx.x) >> 5;
  const int lane = threadIdx.x & 31;
  const int ntile = N >> 6;
  const int m0 = (wave / ntile) << 6;
  const int n0 = (wave % ntile) << 6;
  if (m0 >= M) return;
  const int mrow = lane & 15;
  const int half = lane >> 4;

  v8f acc[4][4] = {};                       // [m-tile][n-tile]

  // A fragment: lane (m) holds K = {8h..8h+7} U {16+8h..16+8h+7}
  const __bf16* arow = A + (size_t)(m0 + mrow) * K + 8 * half;
  const size_t astep = (size_t)16 * K;      // 16 rows
  for (int k0 = 0; k0 < K; k0 += 32) {
    v16bf af[4];
#pragma unroll
    for (int tm = 0; tm < 4; ++tm) {
      const __bf16* ap = arow + (size_t)tm * astep + k0;
      af[tm] = load_frag2(ap, ap + 16);
    }
    __builtin_prefetch(arow + k0 + 512, 0, 0);   // ~16 K-steps ahead
#pragma unroll
    for (int tn = 0; tn < 4; ++tn) {
      // B fragment: lane (n) holds K = {8h..8h+7} U {16+8h..16+8h+7} in Bt.
      const __bf16* bp = Bt + (size_t)(n0 + tn * 16 + mrow) * K + k0 + 8 * half;
      v16bf bf = load_frag2(bp, bp + 16);
      if (tn == 0) __builtin_prefetch(bp + 512, 0, 0);
#pragma unroll
      for (int tm = 0; tm < 4; ++tm)
        acc[tm][tn] = wmma_bf16(af[tm], bf, acc[tm][tn]);
    }
  }
  // C layout: element r -> row 8*half + r, col = lane&15
  const int rbase = half * 8;
#pragma unroll
  for (int tn = 0; tn < 4; ++tn) {
    float bv = bias[n0 + tn * 16 + mrow];
#pragma unroll
    for (int tm = 0; tm < 4; ++tm) {
#pragma unroll
      for (int r = 0; r < 8; ++r) {
        C[(size_t)(m0 + tm * 16 + rbase + r) * N + (n0 + tn * 16 + mrow)] =
            to_out<OutT>(acc[tm][tn][r] + bv);
      }
    }
  }
}

// ---------------------------------------------------------------------------
// Flash-attention (grouped-query). One wave = one (b, kvh, g, 16-row q tile).
// qb: [B*L][2048] bf16, kb: [B*L][512] bf16, vt: [B][8][64][L] bf16,
// ctx: [B*L][2048] bf16.
// ---------------------------------------------------------------------------
__global__ __launch_bounds__(256)
void gqa_flash(const __bf16* __restrict__ qb, const __bf16* __restrict__ kb,
               const __bf16* __restrict__ vt, __bf16* __restrict__ ctx) {
  const int wg   = (blockIdx.x * blockDim.x + threadIdx.x) >> 5;
  const int lane = threadIdx.x & 31;
  const int wIn  = threadIdx.x >> 5;          // wave in block (0..7)

  // wg = ((b*8 + kvh)*4 + g)*128 + qt
  const int qt  = wg & 127;
  int rest      = wg >> 7;
  const int g   = rest & 3;  rest >>= 2;
  const int kvh = rest & 7;
  const int b   = rest >> 3;
  const int qh  = kvh * 4 + g;
  const int l0  = qt << 4;

  const int mrow = lane & 15;
  const int half = lane >> 4;

  __shared__ __bf16 pbuf[8][16 * 32];         // per-wave P staging (8 KB)
  __bf16* pw = pbuf[wIn];

  // Q tile A-fragments for HD=64 (two K-steps of 32).
  const __bf16* qrow =
      qb + (size_t)(b * 2048 + l0 + mrow) * 2048 + qh * 64 + 16 * half;
  const v16bf aq0 = load_frag2(qrow,      qrow + 8);
  const v16bf aq1 = load_frag2(qrow + 32, qrow + 40);

  v8f acc[4] = {};
  float mrun[8], lrun[8];
#pragma unroll
  for (int r = 0; r < 8; ++r) { mrun[r] = -3.0e38f; lrun[r] = 0.0f; }

  const size_t krow0  = (size_t)(b * 2048) * 512 + kvh * 64;
  const size_t vtbase = (size_t)(b * 8 + kvh) * 64 * 2048;

  for (int kt = 0; kt < 64; ++kt) {           // 2048 keys / 32 per chunk
    const int keyb = kt << 5;

    // ---- scores S = Q * K^T (16 x 32) ----
    v8f s0 = {}, s1 = {};
    {
      const __bf16* kp0 = kb + krow0 + (size_t)(keyb + mrow) * 512 + 16 * half;
      const __bf16* kp1 = kp0 + (size_t)16 * 512;
      v16bf b0a = load_frag2(kp0,      kp0 + 8);      // dims 0..31
      v16bf b0b = load_frag2(kp0 + 32, kp0 + 40);     // dims 32..63
      s0 = wmma_bf16(aq0, b0a, s0);
      s0 = wmma_bf16(aq1, b0b, s0);
      v16bf b1a = load_frag2(kp1,      kp1 + 8);
      v16bf b1b = load_frag2(kp1 + 32, kp1 + 40);
      s1 = wmma_bf16(aq0, b1a, s1);
      s1 = wmma_bf16(aq1, b1b, s1);
    }

    // ---- online softmax (fp32). Row r lives on all 16 lanes of one half. ----
    float alpha[8], p0v[8], p1v[8];
#pragma unroll
    for (int r = 0; r < 8; ++r) {
      float x0 = s0[r] * 0.125f;              // 1/sqrt(64)
      float x1 = s1[r] * 0.125f;
      float mx = fmaxf(x0, x1);
      mx = fmaxf(mx, swz_xor<1>(mx));
      mx = fmaxf(mx, swz_xor<2>(mx));
      mx = fmaxf(mx, swz_xor<4>(mx));
      mx = fmaxf(mx, swz_xor<8>(mx));
      float mn = fmaxf(mrun[r], mx);
      float al = __expf(mrun[r] - mn);
      float e0 = __expf(x0 - mn);
      float e1 = __expf(x1 - mn);
      float rs = e0 + e1;
      rs += swz_xor<1>(rs);
      rs += swz_xor<2>(rs);
      rs += swz_xor<4>(rs);
      rs += swz_xor<8>(rs);
      lrun[r] = lrun[r] * al + rs;
      mrun[r] = mn;
      alpha[r] = al; p0v[r] = e0; p1v[r] = e1;
    }
#pragma unroll
    for (int t = 0; t < 4; ++t)
#pragma unroll
      for (int r = 0; r < 8; ++r) acc[t][r] *= alpha[r];

    // ---- stage P (16x32) to LDS row-major, reload as A fragment ----
#pragma unroll
    for (int r = 0; r < 8; ++r) {
      const int row = half * 8 + r;
      pw[row * 32 + mrow]      = f2bf(p0v[r]);
      pw[row * 32 + 16 + mrow] = f2bf(p1v[r]);
    }
    __asm volatile("s_wait_dscnt 0x0" ::: "memory");
    const __bf16* pp = pw + mrow * 32 + 16 * half;
    v16bf pa = load_frag2(pp, pp + 8);

    // ---- ctx += P * V  (V^T rows contiguous over keys) ----
#pragma unroll
    for (int t = 0; t < 4; ++t) {
      const __bf16* vp =
          vt + vtbase + (size_t)(t * 16 + mrow) * 2048 + keyb + 16 * half;
      v16bf vb = load_frag2(vp, vp + 8);
      acc[t] = wmma_bf16(pa, vb, acc[t]);
    }
  }

  // ---- normalize and store context ----
#pragma unroll
  for (int r = 0; r < 8; ++r) {
    const float inv = 1.0f / lrun[r];
    const int row = half * 8 + r;
#pragma unroll
    for (int t = 0; t < 4; ++t) {
      ctx[(size_t)(b * 2048 + l0 + row) * 2048 + qh * 64 + t * 16 + mrow] =
          f2bf(acc[t][r] * inv);
    }
  }
}

// ---------------------------------------------------------------------------
// Host-side orchestration
// ---------------------------------------------------------------------------
extern "C" void kernel_launch(void* const* d_in, const int* in_sizes, int n_in,
                              void* d_out, int out_size, void* d_ws,
                              size_t ws_size, hipStream_t stream) {
  (void)in_sizes; (void)n_in; (void)out_size; (void)ws_size;

  const float* x  = (const float*)d_in[0];
  const float* Wq = (const float*)d_in[1];
  const float* bq = (const float*)d_in[2];
  const float* Wk = (const float*)d_in[3];
  const float* bk = (const float*)d_in[4];
  const float* Wv = (const float*)d_in[5];
  const float* bv = (const float*)d_in[6];
  const float* Wo = (const float*)d_in[7];
  const float* bo = (const float*)d_in[8];
  float* out = (float*)d_out;

  // Workspace carve-up (bf16 elements), ~80 MB total.
  __bf16* p = (__bf16*)d_ws;
  __bf16* xb   = p; p += 8388608;   // x bf16            [4096][2048]
  __bf16* WqT  = p; p += 4194304;   // Wq^T              [2048][2048]
  __bf16* WkT  = p; p += 1048576;   // Wk^T              [512][2048]
  __bf16* WvT  = p; p += 1048576;   // Wv^T              [512][2048]
  __bf16* WoT  = p; p += 4194304;   // Wo^T              [2048][2048]
  __bf16* qb   = p; p += 8388608;   // Q                 [4096][2048]
  __bf16* kb   = p; p += 2097152;   // K                 [4096][512]
  __bf16* vb   = p; p += 2097152;   // V                 [4096][512]
  __bf16* vtb  = p; p += 2097152;   // V^T per head      [2][8][64][2048]
  __bf16* ctxb = p; p += 8388608;   // attention output  [4096][2048]

  // 1) precision conversion (+ weight transpose for B-fragment locality)
  k_cvt  <<<8388608 / 256, 256, 0, stream>>>(x,  xb,  8388608);
  k_cvt_t<<<4194304 / 256, 256, 0, stream>>>(Wq, WqT, 2048, 2048);
  k_cvt_t<<<1048576 / 256, 256, 0, stream>>>(Wk, WkT, 2048, 512);
  k_cvt_t<<<1048576 / 256, 256, 0, stream>>>(Wv, WvT, 2048, 512);
  k_cvt_t<<<4194304 / 256, 256, 0, stream>>>(Wo, WoT, 2048, 2048);

  // 2) Q/K/V projections (WMMA GEMMs, 64x64 wave tiles, 8 waves/block)
  gemm_bf16_wmma<__bf16><<<256, 256, 0, stream>>>(xb, WqT, bq, qb, 4096, 2048, 2048);
  gemm_bf16_wmma<__bf16><<< 64, 256, 0, stream>>>(xb, WkT, bk, kb, 4096,  512, 2048);
  gemm_bf16_wmma<__bf16><<< 64, 256, 0, stream>>>(xb, WvT, bv, vb, 4096,  512, 2048);

  // 3) V transpose for P@V fragment layout
  k_vT<<<2097152 / 256, 256, 0, stream>>>(vb, vtb);

  // 4) grouped-query flash attention (8192 waves)
  gqa_flash<<<1024, 256, 0, stream>>>(qb, kb, vtb, ctxb);

  // 5) output projection -> fp32 d_out
  gemm_bf16_wmma<float><<<256, 256, 0, stream>>>(ctxb, WoT, bo, out, 4096, 2048, 2048);
}
